// Block_53446573032289
// MI455X (gfx1250) — matrix operand order repaired and checked by reading, not compiled, not run on hardware
//
#include <hip/hip_runtime.h>
#include <hip/hip_bf16.h>
#include <cstdint>

// ---------------------------------------------------------------------------
// Transformer encoder block on gfx1250 (CDNA5 / MI455X).
//  * all GEMMs on v_wmma_f32_16x16x32_bf16, fp32 accumulation
//  * bf16 end-to-end operands, converted once
//  * tiles DMA'd into LDS with global_load_async_to_lds_b128 (ASYNCcnt),
//    double-buffered so tile k+1 DMA overlaps tile k WMMAs
//  * softmax row reductions via DPP16 ROW_XMASK (VALU, no LDS round-trips)
// ---------------------------------------------------------------------------

typedef __attribute__((ext_vector_type(4)))  __bf16 v4bf;
typedef __attribute__((ext_vector_type(8)))  float  v8f;
typedef __attribute__((ext_vector_type(8)))  __bf16 v8bf;
typedef __attribute__((ext_vector_type(16))) __bf16 v16bf;

#define SHUF16(lo, hi) __builtin_shufflevector((lo), (hi), 0,1,2,3,4,5,6,7,8,9,10,11,12,13,14,15)

static __device__ __forceinline__ v8f wmma_bf16(v16bf a, v16bf b, v8f c) {
    return __builtin_amdgcn_wmma_f32_16x16x32_bf16(false, a, false, b, (short)0, c, false, false);
}

static __device__ __forceinline__ void async_b128(uint32_t lds, const void* gp) {
    asm volatile("global_load_async_to_lds_b128 %0, %1, off"
                 :: "v"(lds), "v"(gp) : "memory");
}
static __device__ __forceinline__ void wait_async0() {
    asm volatile("s_wait_asynccnt 0x0" ::: "memory");
}
static __device__ __forceinline__ uint32_t lds_off(const void* p) {
    return (uint32_t)(uintptr_t)p;   // low 32 bits of generic shared ptr = LDS byte offset
}

// lane <- lane^MASK within each 16-lane row (DPP16 ROW_XMASK): pure VALU.
template<int MASK>
static __device__ __forceinline__ float xor16(float v) {
    int r = __builtin_amdgcn_update_dpp(0, __float_as_int(v), 0x160 | MASK, 0xf, 0xf, true);
    return __int_as_float(r);
}
static __device__ __forceinline__ float rmax16(float v) {
    v = fmaxf(v, xor16<1>(v));
    v = fmaxf(v, xor16<2>(v));
    v = fmaxf(v, xor16<4>(v));
    v = fmaxf(v, xor16<8>(v));
    return v;
}
static __device__ __forceinline__ float rsum16(float v) {
    v += xor16<1>(v);
    v += xor16<2>(v);
    v += xor16<4>(v);
    v += xor16<8>(v);
    return v;
}

// ======================= f32 -> bf16 convert =======================
__global__ __launch_bounds__(256)
void cvt_kernel(const float* __restrict__ in, __bf16* __restrict__ out, long n)
{
    long i = ((long)blockIdx.x * 256 + threadIdx.x) * 4;
    if (i + 3 < n) {
        float4 v = *(const float4*)(in + i);
        v4bf o; o[0] = (__bf16)v.x; o[1] = (__bf16)v.y; o[2] = (__bf16)v.z; o[3] = (__bf16)v.w;
        *(v4bf*)(out + i) = o;
    }
}

// ======================= GEMM: Y[M,N] = A[M,K](bf16) @ W[N,K](bf16)^T + bias ===========
// EP: 0 = bf16 scatter to [B,12,4096,64]  (Q/K; `scale` applied after bias)
//     1 = bf16 scatter to [B,12,64,4096]  (V transposed)
//     2 = fp32 out = acc + bias + res[M,N]
//     3 = bf16 out = GELU(acc + bias)
#define GBM 128
#define GBN 64
#define GBK 32
#define GLDA 40   // 32 data + 8 pad bf16 -> 80B row stride; 16B-aligned chunks
#define GLDB 40

template<int EP>
__global__ __launch_bounds__(256)
void gemm_kernel(const __bf16* __restrict__ A, const __bf16* __restrict__ W,
                 const float* __restrict__ bias, const float* __restrict__ res,
                 void* __restrict__ outv, int M, int N, int K, float scale)
{
    __shared__ __bf16 As[2][GBM * GLDA];
    __shared__ __bf16 Bs[2][GBN * GLDB];

    const int t     = threadIdx.x;
    const int lane  = t & 31;
    const int wid   = t >> 5;
    const int waveM = wid & 3;
    const int waveN = wid >> 2;
    const int half  = lane >> 4;
    const int l16   = lane & 15;

    const long mBlk = (long)blockIdx.y * GBM;
    const long nBlk = (long)blockIdx.x * GBN;

    auto issue = [&](int kt, int buf) {
        const __bf16* Ag = A + mBlk * (long)K + kt * GBK;
        const uint32_t aL = lds_off(&As[buf][0]);
        #pragma unroll
        for (int j = 0; j < 2; ++j) {                 // 512 chunks / 256 thr
            int idx = t + 256 * j;
            int row = idx >> 2, c = idx & 3;
            async_b128(aL + row * (GLDA * 2) + c * 16, Ag + (long)row * K + c * 8);
        }
        const __bf16* Wg = W + nBlk * (long)K + kt * GBK;
        const uint32_t bL = lds_off(&Bs[buf][0]);
        {                                             // 256 chunks / 256 thr
            int row = t >> 2, c = t & 3;
            async_b128(bL + row * (GLDB * 2) + c * 16, Wg + (long)row * K + c * 8);
        }
    };

    v8f acc[2][2] = {};
    const int nk = K / GBK;
    issue(0, 0);

    for (int kt = 0; kt < nk; ++kt) {
        const int buf = kt & 1;
        wait_async0();
        __syncthreads();
        if (kt + 1 < nk) issue(kt + 1, buf ^ 1);

        v16bf afrag[2], bfrag[2];
        #pragma unroll
        for (int i = 0; i < 2; ++i) {
            int row = waveM * 32 + i * 16 + l16;
            int kb  = half ? 8 : 0;
            const __bf16* p = &As[buf][row * GLDA + kb];
            v8bf lo = *(const v8bf*)p;
            v8bf hi = *(const v8bf*)(p + 16);
            afrag[i] = SHUF16(lo, hi);
        }
        #pragma unroll
        for (int j = 0; j < 2; ++j) {
            int nrow = waveN * 32 + j * 16 + l16;
            int kb   = half ? 8 : 0;
            const __bf16* p = &Bs[buf][nrow * GLDB + kb];
            v8bf lo = *(const v8bf*)p;
            v8bf hi = *(const v8bf*)(p + 16);
            bfrag[j] = SHUF16(lo, hi);
        }
        #pragma unroll
        for (int i = 0; i < 2; ++i)
            #pragma unroll
            for (int j = 0; j < 2; ++j)
                acc[i][j] = wmma_bf16(afrag[i], bfrag[j], acc[i][j]);
    }

    // ---- epilogue (C/D layout: VGPR r, lane l -> M = r + 8*(l>>4), N = l&15)
    #pragma unroll
    for (int i = 0; i < 2; ++i) {
        #pragma unroll
        for (int j = 0; j < 2; ++j) {
            long n  = nBlk + waveN * 32 + j * 16 + l16;
            float bv = bias[n];
            #pragma unroll
            for (int r = 0; r < 8; ++r) {
                long m = mBlk + waveM * 32 + i * 16 + r + 8 * half;
                float v = acc[i][j][r] + bv;
                if constexpr (EP == 0) {
                    long b = m >> 12, s = m & 4095, h = n >> 6, d = n & 63;
                    ((__bf16*)outv)[(((b * 12 + h) * 4096) + s) * 64 + d] = (__bf16)(v * scale);
                } else if constexpr (EP == 1) {
                    long b = m >> 12, s = m & 4095, h = n >> 6, d = n & 63;
                    ((__bf16*)outv)[(((b * 12 + h) * 64) + d) * 4096 + s] = (__bf16)v;
                } else if constexpr (EP == 2) {
                    ((float*)outv)[m * (long)N + n] = v + res[m * (long)N + n];
                } else {
                    float g = 0.5f * v * (1.0f + erff(v * 0.70710678118654752f));
                    ((__bf16*)outv)[m * (long)N + n] = (__bf16)g;
                }
            }
        }
    }
}

// ======================= Flash attention (full, no mask) =======================
// Q,K bf16 [B,12,4096,64]; V bf16 [B,12,64,4096] (transposed); ctx bf16 [B,4096,768].
// 128 thr = 4 waves; 16 query rows/wave; key tiles of 64, double-buffered async DMA.
#define ALDK 72     // 64 data + 8 pad bf16 -> 144B stride
#define APLD 72

__global__ __launch_bounds__(128)
void attn_kernel(const __bf16* __restrict__ Q, const __bf16* __restrict__ Kb,
                 const __bf16* __restrict__ Vb, __bf16* __restrict__ Ctx)
{
    __shared__ __bf16 Ks[2][64 * ALDK];   // [key][d]
    __shared__ __bf16 Vt[2][64 * ALDK];   // [d][key]
    __shared__ __bf16 Ps[4 * 16 * APLD];  // wave-private P tiles

    const int t    = threadIdx.x;
    const int lane = t & 31;
    const int wid  = t >> 5;
    const int half = lane >> 4;
    const int l16  = lane & 15;

    const int  bh   = blockIdx.x;              // b*12 + h
    const long base = (long)bh * 4096 * 64;
    const int  q0   = blockIdx.y * 64 + wid * 16;

    auto issue = [&](int kt0, int buf) {
        const uint32_t kL = lds_off(&Ks[buf][0]);
        const uint32_t vL = lds_off(&Vt[buf][0]);
        #pragma unroll
        for (int j = 0; j < 4; ++j) {               // 512 chunks / 128 thr
            int idx = t + 128 * j;
            int row = idx >> 3, c = idx & 7;
            async_b128(kL + row * (ALDK * 2) + c * 16,
                       Kb + base + (long)(kt0 + row) * 64 + c * 8);
            async_b128(vL + row * (ALDK * 2) + c * 16,
                       Vb + base + (long)row * 4096 + kt0 + c * 8);
        }
    };

    // ---- Q fragments straight from global bf16 (1/sqrt(dk) folded into Q proj)
    v16bf qf[2];
    {
        const __bf16* qrow = Q + base + (long)(q0 + l16) * 64;
        #pragma unroll
        for (int c = 0; c < 2; ++c) {
            int kb = c * 32 + (half ? 8 : 0);
            v8bf lo = *(const v8bf*)(qrow + kb);
            v8bf hi = *(const v8bf*)(qrow + kb + 16);
            qf[c] = SHUF16(lo, hi);
        }
    }

    float mstate[8], lstate[8];
    #pragma unroll
    for (int r = 0; r < 8; ++r) { mstate[r] = -1e30f; lstate[r] = 0.0f; }
    v8f cacc[4] = {};
    __bf16* Pw = &Ps[wid * 16 * APLD];

    issue(0, 0);
    for (int it = 0; it < 64; ++it) {
        const int buf = it & 1;
        wait_async0();
        __syncthreads();
        if (it + 1 < 64) issue((it + 1) * 64, buf ^ 1);

        // ---- scores: S = Q @ K^T
        v8f sacc[4] = {};
        #pragma unroll
        for (int s = 0; s < 4; ++s) {
            #pragma unroll
            for (int c = 0; c < 2; ++c) {
                int key = s * 16 + l16;
                int db  = c * 32 + (half ? 8 : 0);
                const __bf16* p = &Ks[buf][key * ALDK + db];
                v8bf lo = *(const v8bf*)p;
                v8bf hi = *(const v8bf*)(p + 16);
                sacc[s] = wmma_bf16(qf[c], SHUF16(lo, hi), sacc[s]);
            }
        }

        // ---- online softmax (DPP row reductions over 16-lane halves)
        float corr[8];
        #pragma unroll
        for (int r = 0; r < 8; ++r) {
            float v = fmaxf(fmaxf(sacc[0][r], sacc[1][r]), fmaxf(sacc[2][r], sacc[3][r]));
            v = rmax16(v);
            float mnew = fmaxf(mstate[r], v);
            corr[r]   = __expf(mstate[r] - mnew);
            mstate[r] = mnew;
        }
        #pragma unroll
        for (int r = 0; r < 8; ++r) {
            float srow = 0.0f;
            #pragma unroll
            for (int s = 0; s < 4; ++s) {
                float p = __expf(sacc[s][r] - mstate[r]);
                sacc[s][r] = p;
                srow += p;
            }
            srow = rsum16(srow);
            lstate[r] = lstate[r] * corr[r] + srow;
        }
        #pragma unroll
        for (int dts = 0; dts < 4; ++dts)
            #pragma unroll
            for (int r = 0; r < 8; ++r) cacc[dts][r] *= corr[r];

        // ---- stage P as bf16 (wave-private; per-wave LDS ordering covers RAW)
        #pragma unroll
        for (int s = 0; s < 4; ++s)
            #pragma unroll
            for (int r = 0; r < 8; ++r)
                Pw[(r + 8 * half) * APLD + s * 16 + l16] = (__bf16)sacc[s][r];

        // ---- ctx += P @ V
        #pragma unroll
        for (int c = 0; c < 2; ++c) {
            int kb = c * 32 + (half ? 8 : 0);
            const __bf16* pp = &Pw[l16 * APLD + kb];
            v8bf alo = *(const v8bf*)pp;
            v8bf ahi = *(const v8bf*)(pp + 16);
            v16bf af = SHUF16(alo, ahi);
            #pragma unroll
            for (int dts = 0; dts < 4; ++dts) {
                int d   = dts * 16 + l16;
                int kbb = c * 32 + (half ? 8 : 0);
                const __bf16* p = &Vt[buf][d * ALDK + kbb];
                v8bf lo = *(const v8bf*)p;
                v8bf hi = *(const v8bf*)(p + 16);
                cacc[dts] = wmma_bf16(af, SHUF16(lo, hi), cacc[dts]);
            }
        }
    }

    // ---- normalize, write bf16 ctx to [B,4096,768]
    const int h = bh % 12, b = bh / 12;
    #pragma unroll
    for (int dts = 0; dts < 4; ++dts) {
        #pragma unroll
        for (int r = 0; r < 8; ++r) {
            int  q   = q0 + r + 8 * half;
            long row = (long)b * 4096 + q;
            int  col = h * 64 + dts * 16 + l16;
            Ctx[row * 768 + col] = (__bf16)(cacc[dts][r] / lstate[r]);
        }
    }
}

// ======================= LayerNorm (wave per row, D=768, float4) =======================
template<bool DUAL>
__global__ __launch_bounds__(256)
void ln_kernel(const float* __restrict__ X, const float* __restrict__ g,
               const float* __restrict__ bta, float* __restrict__ outF,
               __bf16* __restrict__ outB, int rows)
{
    const int lane = threadIdx.x & 31;
    const int wid  = threadIdx.x >> 5;
    const long row = (long)blockIdx.x * 8 + wid;
    if (row >= rows) return;

    const float* x = X + row * 768;
    float4 vals[6];
    float s1 = 0.0f, s2 = 0.0f;
    #pragma unroll
    for (int j = 0; j < 6; ++j) {
        float4 v = *(const float4*)(x + (lane + 32 * j) * 4);
        vals[j] = v;
        s1 += v.x + v.y + v.z + v.w;
        s2 += v.x * v.x + v.y * v.y + v.z * v.z + v.w * v.w;
    }
    s1 = rsum16(s1);                         // DPP within 16-lane rows
    s2 = rsum16(s2);
    s1 += __shfl_xor(s1, 16, 32);            // single cross-half step
    s2 += __shfl_xor(s2, 16, 32);
    const float mean = s1 * (1.0f / 768.0f);
    const float var  = s2 * (1.0f / 768.0f) - mean * mean;
    const float inv  = rsqrtf(var + 1e-5f);

    #pragma unroll
    for (int j = 0; j < 6; ++j) {
        int c = (lane + 32 * j) * 4;
        float4 gg = *(const float4*)(g + c);
        float4 bb = *(const float4*)(bta + c);
        float4 y;
        y.x = (vals[j].x - mean) * inv * gg.x + bb.x;
        y.y = (vals[j].y - mean) * inv * gg.y + bb.y;
        y.z = (vals[j].z - mean) * inv * gg.z + bb.z;
        y.w = (vals[j].w - mean) * inv * gg.w + bb.w;
        *(float4*)(outF + row * 768 + c) = y;
        if constexpr (DUAL) {
            v4bf o; o[0] = (__bf16)y.x; o[1] = (__bf16)y.y;
            o[2] = (__bf16)y.z; o[3] = (__bf16)y.w;
            *(v4bf*)(outB + row * 768 + c) = o;
        }
    }
}

// ======================= Launch =======================
extern "C" void kernel_launch(void* const* d_in, const int* in_sizes, int n_in,
                              void* d_out, int out_size, void* d_ws, size_t ws_size,
                              hipStream_t stream) {
    (void)in_sizes; (void)n_in; (void)out_size; (void)ws_size;
    const float* x     = (const float*)d_in[0];
    const float* wq    = (const float*)d_in[1];
    const float* bq    = (const float*)d_in[2];
    const float* wk    = (const float*)d_in[3];
    const float* bk    = (const float*)d_in[4];
    const float* wv    = (const float*)d_in[5];
    const float* bv    = (const float*)d_in[6];
    const float* wo    = (const float*)d_in[7];
    const float* bo    = (const float*)d_in[8];
    const float* w1    = (const float*)d_in[9];
    const float* b1    = (const float*)d_in[10];
    const float* w2    = (const float*)d_in[11];
    const float* b2    = (const float*)d_in[12];
    const float* g1    = (const float*)d_in[13];
    const float* beta1 = (const float*)d_in[14];
    const float* g2    = (const float*)d_in[15];
    const float* beta2 = (const float*)d_in[16];

    const long M   = 8192;
    const long MD  = M * 768;        // 6291456
    const long WD  = 768L * 768;     // 589824
    const long WH  = 3072L * 768;    // 2359296

    __bf16* bp = (__bf16*)d_ws;
    auto ab = [&](long n) { __bf16* r = bp; bp += n; return r; };
    __bf16* xb   = ab(MD);
    __bf16* wqb  = ab(WD);
    __bf16* wkb  = ab(WD);
    __bf16* wvb  = ab(WD);
    __bf16* wob  = ab(WD);
    __bf16* w1b  = ab(WH);
    __bf16* w2b  = ab(WH);
    __bf16* qb   = ab(MD);
    __bf16* kb   = ab(MD);
    __bf16* vb   = ab(MD);           // transposed [B,12,64,4096]
    __bf16* ctxb = ab(MD);
    __bf16* n1b  = ab(MD);
    __bf16* hdn  = ab(M * 3072);
    float* fp = (float*)bp;
    float* t1 = fp;  fp += MD;
    float* n1 = fp;  fp += MD;
    float* t2 = fp;
    float* out = (float*)d_out;

    // ---- one-time bf16 conversions
    cvt_kernel<<<dim3(MD / 1024), 256, 0, stream>>>(x,  xb,  MD);
    cvt_kernel<<<dim3(WD / 1024), 256, 0, stream>>>(wq, wqb, WD);
    cvt_kernel<<<dim3(WD / 1024), 256, 0, stream>>>(wk, wkb, WD);
    cvt_kernel<<<dim3(WD / 1024), 256, 0, stream>>>(wv, wvb, WD);
    cvt_kernel<<<dim3(WD / 1024), 256, 0, stream>>>(wo, wob, WD);
    cvt_kernel<<<dim3(WH / 1024), 256, 0, stream>>>(w1, w1b, WH);
    cvt_kernel<<<dim3(WH / 1024), 256, 0, stream>>>(w2, w2b, WH);

    dim3 gQKV(768 / GBN, M / GBM);   // 12 x 64
    dim3 gFF1(3072 / GBN, M / GBM);  // 48 x 64
    dim3 gAttn(24, 4096 / 64);

    // Q (scaled by 1/sqrt(64)), K head-major; V transposed
    gemm_kernel<0><<<gQKV, 256, 0, stream>>>(xb, wqb, bq, nullptr, qb, M, 768, 768, 0.125f);
    gemm_kernel<0><<<gQKV, 256, 0, stream>>>(xb, wkb, bk, nullptr, kb, M, 768, 768, 1.0f);
    gemm_kernel<1><<<gQKV, 256, 0, stream>>>(xb, wvb, bv, nullptr, vb, M, 768, 768, 1.0f);
    attn_kernel<<<gAttn, 128, 0, stream>>>(qb, kb, vb, ctxb);
    // O-proj + residual(x) -> t1 (fp32)
    gemm_kernel<2><<<gQKV, 256, 0, stream>>>(ctxb, wob, bo, x, t1, M, 768, 768, 1.0f);
    ln_kernel<true><<<1024, 256, 0, stream>>>(t1, g1, beta1, n1, n1b, (int)M);
    // FFN1 + GELU -> hdn (bf16)
    gemm_kernel<3><<<gFF1, 256, 0, stream>>>(n1b, w1b, b1, nullptr, hdn, M, 3072, 768, 1.0f);
    // FFN2 + residual(n1) -> t2 (fp32)
    gemm_kernel<2><<<gQKV, 256, 0, stream>>>(hdn, w2b, b2, n1, t2, M, 768, 3072, 1.0f);
    ln_kernel<false><<<1024, 256, 0, stream>>>(t2, g2, beta2, out, nullptr, (int)M);
}
